// Model_72851235275130
// MI455X (gfx1250) — matrix operand order repaired and checked by reading, not compiled, not run on hardware
//
#include <hip/hip_runtime.h>
#include <hip/hip_bf16.h>

// ---------------------------------------------------------------------------
// 2-layer RNN LM for MI455X (gfx1250, wave32, WMMA).
//
// Roofline:
//  * logits = h @ emb^T : 4096 x 256 x 32000 = 67 GFLOP -> bf16 WMMA, f32 acc.
//    emb packed to bf16 B-layout = 16.4 MB -> L2-resident (192 MB L2).
//  * probs output = 524 MB f32 -> ~23 us HBM floor. Two-pass fused softmax
//    (recompute WMMA tiles) avoids ~1 GB of logit spill traffic.
//  * RNN recurrence: B=32,N=256 -> exactly 32 16x16 tiles -> one 32-wave
//    workgroup per layer; h ping-pongs through LDS in WMMA A-layout; Wh+Wy
//    (bf16 B-layout, 128 KB each) live in the 320 KB LDS, preloaded with
//    CDNA5 async global->LDS copies (ASYNCcnt) when available.
//  * tanh on the serial path uses gfx1250's native v_tanh_f32 when available.
//
// WMMA 16x16x32 bf16 register layouts assumed (ISA 7.12.2):
//  A (16x32, MxK): lane l holds row m=l&15; per-lane half q (v=q/2,h=q&1):
//      k = (v>=4?16:0) + (l>=16?8:0) + (v&3)*2 + h
//  B (32x16, KxN): lane l holds col n=l&15; k = (v>=4?16:0) + (l>=16?8:0) + (v&3)*2 + h (same as A)
//  C (16x16 f32): vgpr r, lane l: m = r + (l>=16?8:0), n = l&15
// Packed tiles: 512 contiguous bf16 per tile = [lane][16 halves], so each
// lane loads one contiguous 32 B v16bf.
// ---------------------------------------------------------------------------

typedef __attribute__((ext_vector_type(16))) __bf16 v16bf;
typedef __attribute__((ext_vector_type(8)))  float  v8f;

#if defined(__has_builtin)
#if __has_builtin(__builtin_amdgcn_global_load_async_to_lds_b128)
#define HAVE_ASYNC_LDS 1
#endif
#if __has_builtin(__builtin_amdgcn_s_wait_asynccnt)
#define HAVE_WAIT_ASYNC 1
#endif
#if __has_builtin(__builtin_amdgcn_tanhf)
#define HAVE_TANH 1
#endif
#endif

#define RNN_B 32
#define RNN_T 128
#define RNN_H 256
#define RNN_V 32000
#define KT     8        // K tiles of 32 over H=256
#define NT_H  16        // N tiles of 16 over H=256
#define NT_V  2000      // N tiles of 16 over V=32000
#define TILE_HALVES 512 // 32 lanes * 16 halves

// half-index (lane*16 + v*2 + h) inside a 512-entry A tile for element (m,k)
__device__ __forceinline__ int a_lvh(int m, int k) {
  int l = m + (((k & 15) >= 8) ? 16 : 0);
  int v = ((k & 16) ? 4 : 0) + ((k & 7) >> 1);
  return l * 16 + v * 2 + (k & 1);
}

__device__ __forceinline__ v8f wmma_bf16(v16bf a, v16bf b, v8f c) {
  return __builtin_amdgcn_wmma_f32_16x16x32_bf16(false, a, false, b, (short)0, c,
                                                 false, false);
}

__device__ __forceinline__ float fast_tanh(float x) {
#if HAVE_TANH
  return __builtin_amdgcn_tanhf(x);   // gfx1250 native v_tanh_f32
#else
  return tanhf(x);
#endif
}

// ---- pack a row-major f32 [K=256, N] matrix into B-layout bf16 tiles --------
// dst layout: [N/16][8][512]
__global__ void packB_kernel(const float* __restrict__ src, __bf16* __restrict__ dst,
                             int N, int total) {
  int idx = blockIdx.x * blockDim.x + threadIdx.x;
  if (idx >= total) return;
  int nt  = idx >> 12;          // /4096
  int rem = idx & 4095;
  int kt  = rem >> 9;           // /512
  int e   = rem & 511;
  int l = e >> 4, q = e & 15, v = q >> 1, h = q & 1;
  int k = kt * 32 + ((v & 4) ? 16 : 0) + ((l >= 16) ? 8 : 0) + (v & 3) * 2 + h;
  int n = nt * 16 + (l & 15);
  dst[idx] = (__bf16)src[(size_t)k * N + n];
}

// ---- embedding gather + pack into A-layout bf16 tiles -----------------------
// M index ordered as t*B + b so each timestep owns 2 whole 16-row tiles.
// dst layout: [M/16=256][8][512]
__global__ void gather_packA_kernel(const int* __restrict__ tokens,
                                    const float* __restrict__ emb,
                                    __bf16* __restrict__ dst, int total) {
  int idx = blockIdx.x * blockDim.x + threadIdx.x;
  if (idx >= total) return;
  int mt  = idx >> 12;
  int rem = idx & 4095;
  int kt  = rem >> 9;
  int e   = rem & 511;
  int l = e >> 4, q = e & 15, v = q >> 1, h = q & 1;
  int m = l & 15;
  int k = kt * 32 + ((v & 4) ? 16 : 0) + ((l >= 16) ? 8 : 0) + (v & 3) * 2 + h;
  int M = mt * 16 + m;
  int t = M >> 5, b = M & 31;
  int tok = tokens[b * RNN_T + t];
  dst[idx] = (__bf16)emb[(size_t)tok * RNN_H + k];
}

// ---- GEMM: C[4096,256] = Apacked * Bpacked + bias (row vector) --------------
// one wave per 16x16 C tile; 8 waves / 256-thread block; 512 blocks.
__global__ __launch_bounds__(256) void gemm_xw_kernel(
    const __bf16* __restrict__ Ap, const __bf16* __restrict__ Bp,
    const float* __restrict__ bias, float* __restrict__ C) {
  int gw   = blockIdx.x * 8 + (threadIdx.x >> 5);
  int lane = threadIdx.x & 31;
  int ln   = lane & 15;
  int hi   = lane >> 4;
  int mt = gw >> 4, nt = gw & 15;
  float bv = bias[nt * 16 + ln];
  v8f acc;
#pragma unroll
  for (int r = 0; r < 8; ++r) acc[r] = bv;
#pragma unroll
  for (int kt = 0; kt < KT; ++kt) {
    v16bf a = *(const v16bf*)&Ap[((size_t)mt * KT + kt) * TILE_HALVES + lane * 16];
    v16bf b = *(const v16bf*)&Bp[((size_t)nt * KT + kt) * TILE_HALVES + lane * 16];
    acc = wmma_bf16(a, b, acc);
  }
#pragma unroll
  for (int r = 0; r < 8; ++r) {
    int m = r + hi * 8;
    C[(size_t)(mt * 16 + m) * RNN_H + nt * 16 + ln] = acc[r];
  }
}

// ---- RNN scan: one workgroup, 32 waves; h recirculates through LDS ----------
// h_t = tanh(h_{t-1} @ Wh + xw_t) ; y_t = h_t @ Wy + by
// ypack: y in A-layout bf16 (feeds next GEMM / logits).
// hout (layer 1 only): masked y, row-major f32 [B,T,H] into d_out.
__global__ __launch_bounds__(1024) void scan_kernel(
    const float* __restrict__ xw, const __bf16* __restrict__ WhB,
    const __bf16* __restrict__ WyB, const float* __restrict__ by,
    __bf16* __restrict__ ypack, float* __restrict__ hout,
    const int* __restrict__ lengths) {
  __shared__ __bf16 sWh[NT_H * KT * TILE_HALVES];  // 128 KB
  __shared__ __bf16 sWy[NT_H * KT * TILE_HALVES];  // 128 KB
  __shared__ __bf16 sHA[2 * KT * TILE_HALVES];     // 16 KB (h in A-layout)

  int tid = threadIdx.x;
  // zero initial h (plain ds stores)
  {
    unsigned int* z = (unsigned int*)sHA;
    for (int i = tid; i < 4096; i += 1024) z[i] = 0u;
  }
  // preload weights into LDS: CDNA5 async global->LDS when available
#if HAVE_ASYNC_LDS
  {
    typedef __attribute__((ext_vector_type(4))) int v4i_t;
    typedef __attribute__((address_space(1))) v4i_t gvec_t;  // global int4
    typedef __attribute__((address_space(3))) v4i_t lvec_t;  // LDS int4
    gvec_t* g0 = (gvec_t*)(const void*)WhB;
    gvec_t* g1 = (gvec_t*)(const void*)WyB;
    lvec_t* l0 = (lvec_t*)(void*)&sWh[0];
    lvec_t* l1 = (lvec_t*)(void*)&sWy[0];
#pragma unroll
    for (int it = 0; it < 8; ++it) {   // 8192 int4 per matrix / 1024 thr = 8
      int idx = it * 1024 + tid;
      __builtin_amdgcn_global_load_async_to_lds_b128(g0 + idx, l0 + idx, 0, 0);
      __builtin_amdgcn_global_load_async_to_lds_b128(g1 + idx, l1 + idx, 0, 0);
    }
#if HAVE_WAIT_ASYNC
    __builtin_amdgcn_s_wait_asynccnt(0);
#else
    asm volatile("s_wait_asynccnt 0x0" ::: "memory");
#endif
  }
#else
  {
    const uint4* g0 = (const uint4*)WhB; uint4* s0 = (uint4*)sWh;
    const uint4* g1 = (const uint4*)WyB; uint4* s1 = (uint4*)sWy;
    for (int i = tid; i < 8192; i += 1024) { s0[i] = g0[i]; s1[i] = g1[i]; }
  }
#endif
  __syncthreads();

  int wave = tid >> 5, lane = tid & 31;
  int mi = wave >> 4, ni = wave & 15;    // mi: 0..1 (M tile), ni: 0..15 (N tile)
  int ln = lane & 15, hi = lane >> 4;
  int col = ni * 16 + ln;
  int ktc = col >> 5, kkc = col & 31;    // where column `col` lands in A layout
  float byv = by[col];

  for (int t = 0; t < RNN_T; ++t) {
    // --- hidden: acc = xw_t (+bh already) ; += h_prev @ Wh ---
    v8f acc;
#pragma unroll
    for (int r = 0; r < 8; ++r) {
      int m = r + hi * 8;
      acc[r] = xw[(size_t)(t * 32 + mi * 16 + m) * RNN_H + col];
    }
#pragma unroll
    for (int kt = 0; kt < KT; ++kt) {
      v16bf a = *(const v16bf*)&sHA[(mi * KT + kt) * TILE_HALVES + lane * 16];
      v16bf b = *(const v16bf*)&sWh[(ni * KT + kt) * TILE_HALVES + lane * 16];
      acc = wmma_bf16(a, b, acc);
    }
    __syncthreads();                      // all reads of h_prev done
    // --- tanh (native trans op), write h_t back into LDS in A-layout ---
#pragma unroll
    for (int r = 0; r < 8; ++r) {
      int m = r + hi * 8;
      float hv = fast_tanh(acc[r]);
      sHA[(mi * KT + ktc) * TILE_HALVES + a_lvh(m, kkc)] = (__bf16)hv;
    }
    __syncthreads();                      // h_t visible to all waves
    // --- y_t = h_t @ Wy + by ---
    v8f accy;
#pragma unroll
    for (int r = 0; r < 8; ++r) accy[r] = byv;
#pragma unroll
    for (int kt = 0; kt < KT; ++kt) {
      v16bf a = *(const v16bf*)&sHA[(mi * KT + kt) * TILE_HALVES + lane * 16];
      v16bf b = *(const v16bf*)&sWy[(ni * KT + kt) * TILE_HALVES + lane * 16];
      accy = wmma_bf16(a, b, accy);
    }
#pragma unroll
    for (int r = 0; r < 8; ++r) {
      int m   = r + hi * 8;
      int Mrow = t * 32 + mi * 16 + m;    // M = t*B + b
      int mt  = Mrow >> 4;
      ypack[((size_t)mt * KT + ktc) * TILE_HALVES + a_lvh(m, kkc)] = (__bf16)accy[r];
    }
    if (hout) {
#pragma unroll
      for (int r = 0; r < 8; ++r) {
        int m = r + hi * 8;
        int b = mi * 16 + m;
        float mk = (t < lengths[b]) ? 1.0f : 0.0f;
        hout[(size_t)b * (RNN_T * RNN_H) + (size_t)t * RNN_H + col] = accy[r] * mk;
      }
    }
  }
}

// ---- fused logits + softmax + mask ------------------------------------------
// one block per 16-row M tile (256 blocks), 16 waves sweep V in 16-col chunks.
// pass 1: online rowmax / rowsum (wave-shuffle reductions, cross-wave in LDS);
// pass 2: recompute WMMA tiles, write masked probs.
__global__ __launch_bounds__(512) void logits_softmax_kernel(
    const __bf16* __restrict__ Ap, const __bf16* __restrict__ embB,
    const int* __restrict__ lengths, float* __restrict__ probs) {
  __shared__ float redMax[16][16];
  __shared__ float redSum[16][16];
  __shared__ float fMax[16];
  __shared__ float fScale[16];
  __shared__ float rmask[16];

  int mt = blockIdx.x;
  int tid = threadIdx.x, wave = tid >> 5, lane = tid & 31;
  int ln = lane & 15, hi = lane >> 4;

  if (tid < 16) {
    int M = mt * 16 + tid, t = M >> 5, b = M & 31;
    rmask[tid] = (t < lengths[b]) ? 1.0f : 0.0f;
  }

  v16bf a[KT];
#pragma unroll
  for (int kt = 0; kt < KT; ++kt)
    a[kt] = *(const v16bf*)&Ap[((size_t)mt * KT + kt) * TILE_HALVES + lane * 16];

  float Mr[8], Sr[8];
#pragma unroll
  for (int r = 0; r < 8; ++r) { Mr[r] = -3.0e38f; Sr[r] = 0.0f; }

  for (int c = wave; c < NT_V; c += 16) {
    if (c + 16 < NT_V)
      __builtin_prefetch(&embB[((size_t)(c + 16) * KT) * TILE_HALVES + lane * 16], 0, 0);
    v8f acc;
#pragma unroll
    for (int r = 0; r < 8; ++r) acc[r] = 0.0f;
#pragma unroll
    for (int kt = 0; kt < KT; ++kt) {
      v16bf bt = *(const v16bf*)&embB[((size_t)c * KT + kt) * TILE_HALVES + lane * 16];
      acc = wmma_bf16(a[kt], bt, acc);
    }
#pragma unroll
    for (int r = 0; r < 8; ++r) {
      float x = acc[r];
      float mx = x;
      for (int off = 8; off >= 1; off >>= 1) mx = fmaxf(mx, __shfl_xor(mx, off, 32));
      float Mn = fmaxf(Mr[r], mx);
      float e = __expf(x - Mn);
      float s = e;
      for (int off = 8; off >= 1; off >>= 1) s += __shfl_xor(s, off, 32);
      Sr[r] = Sr[r] * __expf(Mr[r] - Mn) + s;
      Mr[r] = Mn;
    }
  }
  if (ln == 0) {
#pragma unroll
    for (int r = 0; r < 8; ++r) {
      redMax[wave][r + hi * 8] = Mr[r];
      redSum[wave][r + hi * 8] = Sr[r];
    }
  }
  __syncthreads();
  if (tid < 16) {
    float Mf = -3.0e38f;
    for (int w = 0; w < 16; ++w) Mf = fmaxf(Mf, redMax[w][tid]);
    float S = 0.0f;
    for (int w = 0; w < 16; ++w) S += redSum[w][tid] * __expf(redMax[w][tid] - Mf);
    fMax[tid] = Mf;
    fScale[tid] = rmask[tid] / S;
  }
  __syncthreads();

  for (int c = wave; c < NT_V; c += 16) {
    if (c + 16 < NT_V)
      __builtin_prefetch(&embB[((size_t)(c + 16) * KT) * TILE_HALVES + lane * 16], 0, 0);
    v8f acc;
#pragma unroll
    for (int r = 0; r < 8; ++r) acc[r] = 0.0f;
#pragma unroll
    for (int kt = 0; kt < KT; ++kt) {
      v16bf bt = *(const v16bf*)&embB[((size_t)c * KT + kt) * TILE_HALVES + lane * 16];
      acc = wmma_bf16(a[kt], bt, acc);
    }
#pragma unroll
    for (int r = 0; r < 8; ++r) {
      int m = r + hi * 8;
      int M = mt * 16 + m, t = M >> 5, b = M & 31;
      float p = __expf(acc[r] - fMax[m]) * fScale[m];
      probs[(size_t)b * RNN_T * RNN_V + (size_t)t * RNN_V + (size_t)c * 16 + ln] = p;
    }
  }
}

// ---------------------------------------------------------------------------
extern "C" void kernel_launch(void* const* d_in, const int* in_sizes, int n_in,
                              void* d_out, int out_size, void* d_ws, size_t ws_size,
                              hipStream_t stream) {
  (void)in_sizes; (void)n_in; (void)out_size; (void)ws_size;
  const int*   tokens  = (const int*)d_in[0];
  const int*   lengths = (const int*)d_in[1];
  const float* emb     = (const float*)d_in[2];
  const float* Wh0 = (const float*)d_in[3];
  const float* Wx0 = (const float*)d_in[4];
  const float* Wy0 = (const float*)d_in[5];
  const float* bh0 = (const float*)d_in[6];
  const float* by0 = (const float*)d_in[7];
  const float* Wh1 = (const float*)d_in[8];
  const float* Wx1 = (const float*)d_in[9];
  const float* Wy1 = (const float*)d_in[10];
  const float* bh1 = (const float*)d_in[11];
  const float* by1 = (const float*)d_in[12];

  float* probs = (float*)d_out;                                  // [B,T,V]
  float* hout  = (float*)d_out + (size_t)RNN_B * RNN_T * RNN_V;  // [B,T,H]

  // workspace carve-up (~32 MB total)
  char* w = (char*)d_ws;
  auto take = [&](size_t bytes) {
    char* p = w;
    w += (bytes + 255) & ~(size_t)255;
    return p;
  };
  __bf16* embB   = (__bf16*)take((size_t)NT_V * KT * TILE_HALVES * 2);  // 16.4 MB
  __bf16* WxB0   = (__bf16*)take((size_t)NT_H * KT * TILE_HALVES * 2);
  __bf16* WhB0   = (__bf16*)take((size_t)NT_H * KT * TILE_HALVES * 2);
  __bf16* WyB0   = (__bf16*)take((size_t)NT_H * KT * TILE_HALVES * 2);
  __bf16* WxB1   = (__bf16*)take((size_t)NT_H * KT * TILE_HALVES * 2);
  __bf16* WhB1   = (__bf16*)take((size_t)NT_H * KT * TILE_HALVES * 2);
  __bf16* WyB1   = (__bf16*)take((size_t)NT_H * KT * TILE_HALVES * 2);
  __bf16* xpack  = (__bf16*)take((size_t)256 * KT * TILE_HALVES * 2);   // 2 MB
  __bf16* yspack = (__bf16*)take((size_t)256 * KT * TILE_HALVES * 2);
  __bf16* h1pack = (__bf16*)take((size_t)256 * KT * TILE_HALVES * 2);
  float*  xw0    = (float*)take((size_t)RNN_B * RNN_T * RNN_H * 4);     // 4 MB
  float*  xw1    = (float*)take((size_t)RNN_B * RNN_T * RNN_H * 4);

  // 1) pack embedding (B layout, K=H, N=V) and the six weight matrices
  {
    int totalE = NT_V * KT * TILE_HALVES;   // 8,192,000
    packB_kernel<<<(totalE + 255) / 256, 256, 0, stream>>>(emb, embB, RNN_V, totalE);
    int totalW = NT_H * KT * TILE_HALVES;   // 65,536
    int gw = (totalW + 255) / 256;
    packB_kernel<<<gw, 256, 0, stream>>>(Wx0, WxB0, RNN_H, totalW);
    packB_kernel<<<gw, 256, 0, stream>>>(Wh0, WhB0, RNN_H, totalW);
    packB_kernel<<<gw, 256, 0, stream>>>(Wy0, WyB0, RNN_H, totalW);
    packB_kernel<<<gw, 256, 0, stream>>>(Wx1, WxB1, RNN_H, totalW);
    packB_kernel<<<gw, 256, 0, stream>>>(Wh1, WhB1, RNN_H, totalW);
    packB_kernel<<<gw, 256, 0, stream>>>(Wy1, WyB1, RNN_H, totalW);
  }
  // 2) x = emb[tokens], packed into A layout (M = t*B + b)
  {
    int totalA = 256 * KT * TILE_HALVES;    // 1,048,576
    gather_packA_kernel<<<(totalA + 255) / 256, 256, 0, stream>>>(tokens, emb, xpack,
                                                                  totalA);
  }
  // 3) layer 0: xw0 = x @ Wx0 + bh0 ; scan
  gemm_xw_kernel<<<512, 256, 0, stream>>>(xpack, WxB0, bh0, xw0);
  scan_kernel<<<1, 1024, 0, stream>>>(xw0, WhB0, WyB0, by0, yspack, nullptr, nullptr);
  // 4) layer 1: xw1 = ys0 @ Wx1 + bh1 ; scan (also writes masked h to d_out)
  gemm_xw_kernel<<<512, 256, 0, stream>>>(yspack, WxB1, bh1, xw1);
  scan_kernel<<<1, 1024, 0, stream>>>(xw1, WhB1, WyB1, by1, h1pack, hout, lengths);
  // 5) logits + softmax + mask -> probs
  logits_softmax_kernel<<<256, 512, 0, stream>>>(h1pack, embB, lengths, probs);
}
